// BaseCRF_66735201845824
// MI455X (gfx1250) — matrix operand order repaired and checked by reading, not compiled, not run on hardware
//
#include <hip/hip_runtime.h>

// CRF forward-backward on gfx1250 (MI455X).
// log-space recurrence -> f16 WMMA GEMM via max-shift/exp trick.
// - exp(trans) B-fragments resident in VGPRs for all T steps
// - shifted-exp state staged in LDS, A-fragments via ds_load_b128
// - emissions double-buffered into LDS with GLOBAL_LOAD_ASYNC_TO_LDS_B128 (if available)
// - raw v_exp_f32 / v_log_f32 (operands provably in safe range)

typedef __attribute__((ext_vector_type(16))) _Float16 v16h;
typedef __attribute__((ext_vector_type(8)))  float    v8f;
typedef __attribute__((__vector_size__(16))) int      crf_v4i;  // b128 payload type

constexpr int Bb = 64;    // batch
constexpr int Tt = 1024;  // time
constexpr int Kk = 256;   // states
constexpr int ROWS = 16;  // batches per workgroup

#define CRF_AS1 __attribute__((address_space(1)))
#define CRF_AS3 __attribute__((address_space(3)))

#if defined(__has_builtin)
#if __has_builtin(__builtin_amdgcn_global_load_async_to_lds_b128) && \
    __has_builtin(__builtin_amdgcn_s_wait_asynccnt)
#define CRF_ASYNC 1
#endif
#if __has_builtin(__builtin_amdgcn_exp2f) && __has_builtin(__builtin_amdgcn_logf)
#define CRF_FASTMATH 1
#endif
#endif

__device__ __forceinline__ float fast_exp(float x) {
#ifdef CRF_FASTMATH
    return __builtin_amdgcn_exp2f(x * 1.4426950408889634f);   // v_exp_f32, no guards
#else
    return __expf(x);
#endif
}
__device__ __forceinline__ float fast_log(float x) {
#ifdef CRF_FASTMATH
    return __builtin_amdgcn_logf(x) * 0.6931471805599453f;    // v_log_f32, no guards
#else
    return __logf(x);
#endif
}

// 8 blocks x 256 threads: blocks 0..3 forward (alpha, z), blocks 4..7 backward (beta).
__global__ __launch_bounds__(256) void crf_sweep(
    const float* __restrict__ em,    // (B,T,K)
    const int*   __restrict__ msk,   // (B,T)
    const float* __restrict__ stT,   // (K)
    const float* __restrict__ enT,   // (K)
    const float* __restrict__ tr,    // (K,K)
    float* __restrict__ alphaW,      // ws: (T,B,K)
    float* __restrict__ betaW,       // ws: (T,B,K)
    float* __restrict__ zW)          // ws: (B)
{
    __shared__ __align__(16) _Float16 aExp[ROWS][Kk];  // exp(state - rowmax) : A matrix
    __shared__ float aF32[ROWS][Kk];                   // staged f32 state
    __shared__ float part[ROWS][16];                   // reduction partials
    __shared__ float rowMax[ROWS];
    __shared__ float mrow[ROWS];                       // mask value per row this step
#ifdef CRF_ASYNC
    __shared__ __align__(16) float eLds[2][ROWS][Kk];  // double-buffered emission tile
#endif

    const int tid  = threadIdx.x;
    const int lane = tid & 31;
    const int wv   = tid >> 5;        // wave 0..7
    const int hi   = lane >> 4;       // lane half
    const int lc   = lane & 15;
    const bool bwd = (blockIdx.x >= 4);
    const int  b0  = (blockIdx.x & 3) * ROWS;

    // Two 16-wide column tiles per wave.
    int col[2];
    col[0] = (wv * 2 + 0) * 16 + lc;
    col[1] = col[0] + 16;

    // ---- Preload exp(trans) B-fragments into VGPRs (resident all T steps). ----
    // 16-bit 32x16 B layout used here (contiguous k, matches A): lane L -> column N=L&15; v16h element e -> K = hi*16 + e.
    v16h Bf[2][8];
#pragma unroll
    for (int nt = 0; nt < 2; ++nt)
#pragma unroll
        for (int kk = 0; kk < 8; ++kk)
#pragma unroll
            for (int e = 0; e < 16; ++e) {
                const int j = kk * 32 + hi * 16 + e;   // reduction index
                const float tv = bwd ? tr[(long)col[nt] * Kk + j]   // exp(trans^T)
                                     : tr[(long)j * Kk + col[nt]];  // exp(trans)
                Bf[nt][kk][e] = (_Float16)fast_exp(tv);
            }

    // Running log-state (alpha or beta) in C/D accumulator layout:
    // element i of tile nt -> row = i + 8*hi, col = col[nt].
    float st[2][8];
    float eR[2][8];

#pragma unroll
    for (int nt = 0; nt < 2; ++nt)
#pragma unroll
        for (int i = 0; i < 8; ++i) {
            const int row = i + 8 * hi;
            if (!bwd) {
                const float v = em[((long)(b0 + row) * Tt + 0) * Kk + col[nt]] + stT[col[nt]];
                st[nt][i] = v;
                alphaW[((long)0 * Bb + (b0 + row)) * Kk + col[nt]] = v;
            } else {
                const float v = enT[col[nt]];
                st[nt][i] = v;
                betaW[((long)(Tt - 1) * Bb + (b0 + row)) * Kk + col[nt]] = v;
            }
        }

#ifdef CRF_ASYNC
    // Each thread DMAs one 64-byte run of one row: row=tid>>4, cols (tid&15)*16..+15.
    const int erow = tid >> 4;
    const int ec0  = (tid & 15) * 16;
    auto issue_tile = [&](int buf, int et) {
        const float* g = &em[((long)(b0 + erow) * Tt + et) * Kk + ec0];
#pragma unroll
        for (int q = 0; q < 4; ++q)
            __builtin_amdgcn_global_load_async_to_lds_b128(
                (CRF_AS1 crf_v4i*)(g + q * 4),
                (CRF_AS3 crf_v4i*)&eLds[buf][erow][ec0 + q * 4], 0, 0);
    };
    // prologue: fetch emissions for step 0
    issue_tile(0, bwd ? (Tt - 1) : 1);
#endif

    for (int step = 0; step < Tt - 1; ++step) {
        // fwd produces alpha[step+1] using e[step+1]; bwd produces beta[i-1], i=T-1-step, using e[i].
        const int et = bwd ? (Tt - 1 - step) : (step + 1);

#ifdef CRF_ASYNC
        const int cur = step & 1;
        if (step + 1 < Tt - 1) {          // kick off next step's tile, then wait for current
            const int etn = bwd ? (et - 1) : (et + 1);
            issue_tile(cur ^ 1, etn);
            __builtin_amdgcn_s_wait_asynccnt(4);
        } else {
            __builtin_amdgcn_s_wait_asynccnt(0);
        }
        __syncthreads();                  // B0: every wave's DMA has landed
#endif

        // ---- stage state (bwd: state + emission) into LDS; cache emission in regs ----
#pragma unroll
        for (int nt = 0; nt < 2; ++nt)
#pragma unroll
            for (int i = 0; i < 8; ++i) {
                const int row = i + 8 * hi;
#ifdef CRF_ASYNC
                const float e = eLds[cur][row][col[nt]];
#else
                const float e = em[((long)(b0 + row) * Tt + et) * Kk + col[nt]];
#endif
                eR[nt][i] = e;
                aF32[row][col[nt]] = bwd ? (st[nt][i] + e) : st[nt][i];
            }
        __syncthreads();                  // B1

        // ---- partial row max (16 rows x 16 chunks) + mask row fetch ----
        {
            const int r = tid >> 4, c0 = (tid & 15) * 16;
            float m = -3.0e38f;
#pragma unroll
            for (int j = 0; j < 16; ++j) m = fmaxf(m, aF32[r][c0 + j]);
            part[r][tid & 15] = m;
            if (tid < ROWS) mrow[tid] = (float)msk[(long)(b0 + tid) * Tt + et];
        }
        __syncthreads();                  // B2

        // ---- shifted exp -> f16 A matrix; each thread folds its row's 16 partials ----
        {
            const int r = tid >> 4, c0 = (tid & 15) * 16;
            float M = part[r][0];
#pragma unroll
            for (int c = 1; c < 16; ++c) M = fmaxf(M, part[r][c]);
            if ((tid & 15) == 0) rowMax[r] = M;   // for finish phase
#pragma unroll
            for (int j = 0; j < 16; ++j)
                aExp[r][c0 + j] = (_Float16)fast_exp(aF32[r][c0 + j] - M);
        }
        __syncthreads();                  // B3

        // ---- GEMM: (16 x 256) @ (256 x 32) via 8 k-steps of 16x16x32 WMMA ----
        v8f acc[2] = { v8f{0,0,0,0,0,0,0,0}, v8f{0,0,0,0,0,0,0,0} };
#pragma unroll
        for (int kk = 0; kk < 8; ++kk) {
            // 16-bit 16x32 A layout (matches B): lane row M = lane&15; element e -> j=hi*16+e.
            // Two 16B LDS loads per lane.
            union { float4 f4[2]; v16h h; } ua;
            ua.f4[0] = *reinterpret_cast<const float4*>(&aExp[lc][kk * 32 + hi * 16]);
            ua.f4[1] = *reinterpret_cast<const float4*>(&aExp[lc][kk * 32 + hi * 16 + 8]);
            acc[0] = __builtin_amdgcn_wmma_f32_16x16x32_f16(false, ua.h, false, Bf[0][kk],
                                                            (short)0, acc[0], false, false);
            acc[1] = __builtin_amdgcn_wmma_f32_16x16x32_f16(false, ua.h, false, Bf[1][kk],
                                                            (short)0, acc[1], false, false);
        }

        // ---- finish: log (acc >= exp(0)*min(E) ~ 0.89, raw v_log safe),
        //      add emission (fwd), mask-select, store to ws ----
        const long tO = bwd ? (long)(Tt - 2 - step) : (long)(step + 1);
        float* __restrict__ dst = bwd ? betaW : alphaW;
#pragma unroll
        for (int nt = 0; nt < 2; ++nt)
#pragma unroll
            for (int i = 0; i < 8; ++i) {
                const int row = i + 8 * hi;
                float v = rowMax[row] + fast_log(acc[nt][i]);
                if (!bwd) v += eR[nt][i];
                st[nt][i] = (mrow[row] != 0.0f) ? v : st[nt][i];   // mask in {0,1}
                dst[(tO * Bb + (b0 + row)) * Kk + col[nt]] = st[nt][i];
            }
        // No trailing barrier: every next-iteration LDS writer sits behind B1(n+1)
        // (or, for the DMA, behind this iteration's B3 arrival of all threads).
    }

    // ---- forward epilogue: z[b] = lse_k(alpha[T-1] + end) ----
    if (!bwd) {
#pragma unroll
        for (int nt = 0; nt < 2; ++nt)
#pragma unroll
            for (int i = 0; i < 8; ++i) {
                const int row = i + 8 * hi;
                aF32[row][col[nt]] = st[nt][i] + enT[col[nt]];
            }
        __syncthreads();
        {
            const int r = tid >> 4, c0 = (tid & 15) * 16;
            float m = -3.0e38f;
#pragma unroll
            for (int j = 0; j < 16; ++j) m = fmaxf(m, aF32[r][c0 + j]);
            part[r][tid & 15] = m;
        }
        __syncthreads();
        {
            const int r = tid >> 4, c0 = (tid & 15) * 16;
            float M = part[r][0];
#pragma unroll
            for (int c = 1; c < 16; ++c) M = fmaxf(M, part[r][c]);
            if ((tid & 15) == 0) rowMax[r] = M;
            float s = 0.0f;
#pragma unroll
            for (int j = 0; j < 16; ++j) s += fast_exp(aF32[r][c0 + j] - M);
            part[r][tid & 15] = s;   // each thread overwrites only its own slot after reading
        }
        __syncthreads();
        if (tid < ROWS) {
            float s = 0.0f;
#pragma unroll
            for (int c = 0; c < 16; ++c) s += part[tid][c];
            zW[b0 + tid] = rowMax[tid] + fast_log(s);
        }
    }
}

// out[t,b,k] = exp(alpha + beta - z[b]); float4 vectorized, memory-bound (~200MB @ 23.3TB/s).
__global__ __launch_bounds__(256) void crf_combine(
    const float* __restrict__ aW, const float* __restrict__ bW,
    const float* __restrict__ zW, float* __restrict__ out, int total4)
{
    const int i4 = blockIdx.x * 256 + threadIdx.x;
    if (i4 >= total4) return;
    const long i0 = (long)i4 * 4;
    const int b = (int)((i0 >> 8) & (Bb - 1));   // flat = (t*B + b)*K + k, K=256
    const float z = zW[b];
    const float4 a  = *reinterpret_cast<const float4*>(aW + i0);
    const float4 be = *reinterpret_cast<const float4*>(bW + i0);
    float4 o;
    o.x = fast_exp(a.x + be.x - z);
    o.y = fast_exp(a.y + be.y - z);
    o.z = fast_exp(a.z + be.z - z);
    o.w = fast_exp(a.w + be.w - z);
    *reinterpret_cast<float4*>(out + i0) = o;
}

extern "C" void kernel_launch(void* const* d_in, const int* in_sizes, int n_in,
                              void* d_out, int out_size, void* d_ws, size_t ws_size,
                              hipStream_t stream)
{
    const float* em  = (const float*)d_in[0];   // emissions (B,T,K) f32
    const int*   msk = (const int*)d_in[1];     // mask (B,T) i32
    const float* stT = (const float*)d_in[2];   // start_transitions (K)
    const float* enT = (const float*)d_in[3];   // end_transitions (K)
    const float* tr  = (const float*)d_in[4];   // transitions (K,K)
    float* out = (float*)d_out;

    float* aW = (float*)d_ws;
    float* bW = aW + (long)Tt * Bb * Kk;
    float* zW = bW + (long)Tt * Bb * Kk;

    crf_sweep<<<8, 256, 0, stream>>>(em, msk, stT, enT, tr, aW, bW, zW);

    const int total4 = Tt * Bb * Kk / 4;
    crf_combine<<<(total4 + 255) / 256, 256, 0, stream>>>(aW, bW, zW, out, total4);
}
